// CausalSelfAttention_25769803776716
// MI455X (gfx1250) — matrix operand order repaired and checked by reading, not compiled, not run on hardware
//
#include <hip/hip_runtime.h>

// ---------------------------------------------------------------------------
// CDNA5 (gfx1250) fused causal self-attention, bf16 WMMA, wave32.
// Data movement: TDM (tensor_load_to_lds, TENSORcnt) for attention K/V tiles,
// global_load_async_to_lds_b128 (ASYNCcnt) for GEMM weight tiles.
// B=2, T=2048, C=2048, H=16 heads, G=4 kv-groups, D=128, F=(16+8)*128=3072
// ---------------------------------------------------------------------------

typedef __bf16 bf16;
typedef __attribute__((ext_vector_type(16))) __bf16 v16bf;
typedef __attribute__((ext_vector_type(8)))  __bf16 v8bf;
typedef __attribute__((ext_vector_type(8)))  float  v8f;
typedef unsigned int u32;
typedef __attribute__((ext_vector_type(4))) unsigned int u32x4;
typedef __attribute__((ext_vector_type(4))) int i32x4;
typedef __attribute__((ext_vector_type(8))) int i32x8;

#define B_    2
#define T_    2048
#define C_    2048
#define NH    16
#define NG    4
#define HD    128
#define F_    3072
#define NSLOT 24          // F_/128 : 24 (g,s) slots, s in [0,6): 0-3=q, 4=k, 5=v
#define QSCALE 0.08838834764831845f   // 1/sqrt(128)

// ---- CDNA5 async global->LDS (ASYNCcnt) ------------------------------------
__device__ __forceinline__ void async_copy16(void* lds_dst, const void* gsrc) {
  asm volatile("global_load_async_to_lds_b128 %0, %1, off"
               :: "v"((u32)(unsigned long long)lds_dst), "v"(gsrc)
               : "memory");
}
__device__ __forceinline__ void wait_async0() {
  asm volatile("s_wait_asynccnt 0" ::: "memory");
}

// ---- CDNA5 Tensor Data Mover: 2D tile global->LDS (TENSORcnt) --------------
// D# per ISA ch.8: group0 = {count/type/addr}, group1 = {dims/strides}.
// dims/strides/tiles in elements (data_size = 2 bytes).
__device__ __forceinline__ void tdm_load_2d(u32 lds_off, const void* gsrc,
                                            u32 dim0, u32 dim1, u32 stride0,
                                            u32 tile0, u32 tile1) {
  const unsigned long long ga = (unsigned long long)gsrc;
  u32x4 g0;
  g0.x = 1u;                                // count=1, user descriptor
  g0.y = lds_off;                           // lds_addr (bytes)
  g0.z = (u32)ga;                           // global_addr[31:0]
  g0.w = (u32)(ga >> 32) | (2u << 30);      // global_addr[56:32] | type=2
  i32x8 g1;
  g1[0] = (int)(1u << 16);                  // data_size=1 -> 2 bytes
  g1[1] = (int)((dim0 & 0xffffu) << 16);    // tensor_dim0[15:0]
  g1[2] = (int)((dim0 >> 16) | ((dim1 & 0xffffu) << 16));
  g1[3] = (int)((dim1 >> 16) | (tile0 << 16));    // tile_dim0
  g1[4] = (int)(tile1 & 0xffffu);                 // tile_dim1 (tile_dim2=0)
  g1[5] = (int)stride0;                           // tensor_dim0_stride[31:0]
  g1[6] = 0;
  g1[7] = 0;
  const i32x4 gz = {0, 0, 0, 0};
#if defined(__clang_major__) && __clang_major__ >= 23
  const i32x8 gz8 = {0, 0, 0, 0, 0, 0, 0, 0};
  __builtin_amdgcn_tensor_load_to_lds(g0, g1, gz, gz, gz8, 0);
#else
  __builtin_amdgcn_tensor_load_to_lds(g0, g1, gz, gz, 0);
#endif
}
__device__ __forceinline__ void wait_tensor0() {
  __builtin_amdgcn_s_wait_tensorcnt(0);
}

// ---- WMMA wrapper ----------------------------------------------------------
__device__ __forceinline__ v8f wmma_bf16(v16bf a, v16bf b, v8f c) {
  return __builtin_amdgcn_wmma_f32_16x16x32_bf16(false, a, false, b,
                                                 (short)0, c, false, false);
}

// A fragment: 16x32 bf16, source row-major [m][k], leading dim ld (elements).
__device__ __forceinline__ v16bf ldA16x32(const bf16* p, int ld) {
  const int lane = threadIdx.x & 31;
  const bf16* r = p + (lane & 15) * (size_t)ld + (lane >> 4) * 8;
  v8bf lo = *(const v8bf*)(r);
  v8bf hi = *(const v8bf*)(r + 16);
  return __builtin_shufflevector(lo, hi, 0,1,2,3,4,5,6,7,8,9,10,11,12,13,14,15);
}

// B fragment: 32x16 bf16 (KxN), source stored as [n][k] row-major, lead dim ld.
__device__ __forceinline__ v16bf ldB32x16(const bf16* p, int ld) {
  const int lane = threadIdx.x & 31;
  const bf16* r = p + (lane & 15) * (size_t)ld + (lane >> 4) * 8;
  v8bf lo = *(const v8bf*)(r);
  v8bf hi = *(const v8bf*)(r + 16);
  return __builtin_shufflevector(lo, hi, 0,1,2,3,4,5,6,7,8,9,10,11,12,13,14,15);
}

// ---- fp32 -> bf16 convert --------------------------------------------------
__global__ void cvt_f32_bf16(const float* __restrict__ src,
                             bf16* __restrict__ dst, int n) {
  int i = blockIdx.x * blockDim.x + threadIdx.x;
  if (i < n) dst[i] = (bf16)src[i];
}

// ---- Kernel 1: QKV GEMM + RoPE + scatter into q / k / v^T ------------------
__global__ void __launch_bounds__(128)
qkv_rope_kernel(const bf16* __restrict__ xb, const bf16* __restrict__ wab,
                const float* __restrict__ cosb, const float* __restrict__ sinb,
                bf16* __restrict__ qb, bf16* __restrict__ kb,
                bf16* __restrict__ vtb) {
  __shared__ __align__(32) bf16 Bt[2][128 * 32];

  const int wid  = threadIdx.x >> 5;
  const int lane = threadIdx.x & 31;
  const int mrow = blockIdx.x * 64 + wid * 16;    // row in [0, B*T)
  const int slot = blockIdx.y;                    // 0..23
  const int g = slot / 6, s = slot % 6;

  const bf16* aBase = xb  + (size_t)mrow * C_;
  const bf16* bBase = wab + (size_t)(slot * 128) * C_;

  auto stageB = [&](int k0, int buf) {
#pragma unroll
    for (int it = 0; it < 4; ++it) {
      const int chunk = threadIdx.x + 128 * it;   // 512 x 16B chunks
      const int row = chunk >> 2;
      const int off = (chunk & 3) * 8;
      async_copy16(&Bt[buf][row * 32 + off],
                   bBase + (size_t)row * C_ + k0 + off);
    }
  };

  const v8f zero = {0.f,0.f,0.f,0.f,0.f,0.f,0.f,0.f};
  v8f acc[8];
#pragma unroll
  for (int j = 0; j < 8; ++j) acc[j] = zero;

  stageB(0, 0);
  for (int k0 = 0; k0 < C_; k0 += 32) {
    const int buf = (k0 >> 5) & 1;
    wait_async0();
    __syncthreads();
    if (k0 + 32 < C_) stageB(k0 + 32, buf ^ 1);
    if (k0 + 32 < C_) __builtin_prefetch(aBase + k0 + 32, 0, 0);
    // batch all fragment loads, then run WMMAs back-to-back
    v16bf a = ldA16x32(aBase + k0, C_);
    v16bf bfr[8];
#pragma unroll
    for (int j = 0; j < 8; ++j) bfr[j] = ldB32x16(&Bt[buf][(j * 16) * 32], 32);
#pragma unroll
    for (int j = 0; j < 8; ++j) acc[j] = wmma_bf16(a, bfr[j], acc[j]);
  }

  const int b  = mrow >> 11;          // / T_
  const int t0 = mrow & (T_ - 1);
  const int hi = lane >> 4, col = lane & 15;

  if (s < 5) {
    const float scale = (s < 4) ? QSCALE : 1.0f;
    bf16* dst = (s < 4)
        ? qb + (size_t)(b * NH + g * 4 + s) * T_ * HD
        : kb + (size_t)(b * NG + g) * T_ * HD;
#pragma unroll
    for (int j = 0; j < 4; ++j) {
#pragma unroll
      for (int e = 0; e < 8; ++e) {
        const int t  = t0 + e + 8 * hi;
        const int d1 = j * 16 + col;
        const int d2 = d1 + 64;
        const float c1 = cosb[t * HD + d1], s1 = sinb[t * HD + d1];
        const float c2 = cosb[t * HD + d2], s2 = sinb[t * HD + d2];
        const float x1 = acc[j][e], x2 = acc[j + 4][e];
        dst[(size_t)t * HD + d1] = (bf16)((x1 * c1 - x2 * s1) * scale);
        dst[(size_t)t * HD + d2] = (bf16)((x2 * c2 + x1 * s2) * scale);
      }
    }
  } else {
    bf16* dst = vtb + (size_t)(b * NG + g) * HD * T_;
#pragma unroll
    for (int j = 0; j < 8; ++j) {
      const int d = j * 16 + col;
#pragma unroll
      for (int e = 0; e < 8; ++e) {
        const int t = t0 + e + 8 * hi;
        dst[(size_t)d * T_ + t] = (bf16)acc[j][e];
      }
    }
  }
}

// ---- Kernel 2: causal flash attention --------------------------------------
// grid = (T/64, NH, B), block = 128 (4 waves). Each wave: 16 q rows, full D.
// K (32x128) and V^T (128x32) tiles DMA'd by the TDM, double-buffered.
__global__ void __launch_bounds__(128)
attn_kernel(const bf16* __restrict__ qb, const bf16* __restrict__ kb,
            const bf16* __restrict__ vtb, bf16* __restrict__ yb) {
  __shared__ __align__(32) bf16 Kt[2][32 * 128];   // [t][d]
  __shared__ __align__(32) bf16 Vt[2][128 * 32];   // [d][t]
  __shared__ __align__(16) bf16 Pl[4][16 * 32];    // per-wave P staging

  const int wid  = threadIdx.x >> 5;
  const int lane = threadIdx.x & 31;
  const int hi = lane >> 4, col = lane & 15;
  const int q0 = blockIdx.x * 64;
  const int m0 = q0 + wid * 16;                    // q-row base (== t)
  const int h = blockIdx.y, b = blockIdx.z, g = h >> 2;

  const bf16* qbase = qb  + ((size_t)(b * NH + h) * T_ + m0) * HD;
  const bf16* kbase = kb  + (size_t)(b * NG + g) * T_ * HD;
  const bf16* vbase = vtb + (size_t)(b * NG + g) * HD * T_;

  // one TDM descriptor pair per KV step, issued by wave 0 only
  auto stageKV = [&](int kv0, int buf) {
    tdm_load_2d((u32)(unsigned long long)&Kt[buf][0],
                kbase + (size_t)kv0 * HD,
                /*dim0=*/HD, /*dim1=*/T_ - kv0, /*stride0=*/HD,
                /*tile0=*/HD, /*tile1=*/32);
    tdm_load_2d((u32)(unsigned long long)&Vt[buf][0],
                vbase + kv0,
                /*dim0=*/T_, /*dim1=*/HD, /*stride0=*/T_,
                /*tile0=*/32, /*tile1=*/HD);
  };

  v16bf qf[4];
#pragma unroll
  for (int dc = 0; dc < 4; ++dc) qf[dc] = ldA16x32(qbase + dc * 32, HD);

  const v8f zero = {0.f,0.f,0.f,0.f,0.f,0.f,0.f,0.f};
  v8f o[8];
  float m[8], l[8];
#pragma unroll
  for (int j = 0; j < 8; ++j) o[j] = zero;
#pragma unroll
  for (int e = 0; e < 8; ++e) { m[e] = -1e30f; l[e] = 0.f; }

  const int kv_end = q0 + 64;          // block-uniform causal bound
  if (wid == 0) stageKV(0, 0);
  for (int kv0 = 0; kv0 < kv_end; kv0 += 32) {
    const int buf = (kv0 >> 5) & 1;
    if (wid == 0) wait_tensor0();
    __syncthreads();
    if (wid == 0 && kv0 + 32 < kv_end) stageKV(kv0 + 32, buf ^ 1);

    // S = Q * K^T  (16 x 32 scores): batch frag loads, then WMMAs
    v8f s0 = zero, s1 = zero;
    v16bf bk[4];
#pragma unroll
    for (int dc = 0; dc < 4; ++dc) bk[dc] = ldB32x16(&Kt[buf][dc * 32], 128);
#pragma unroll
    for (int dc = 0; dc < 4; ++dc) s0 = wmma_bf16(qf[dc], bk[dc], s0);
#pragma unroll
    for (int dc = 0; dc < 4; ++dc)
      bk[dc] = ldB32x16(&Kt[buf][16 * 128 + dc * 32], 128);
#pragma unroll
    for (int dc = 0; dc < 4; ++dc) s1 = wmma_bf16(qf[dc], bk[dc], s1);

    // causal mask + online softmax (row spread over 16 lanes per half-wave)
#pragma unroll
    for (int e = 0; e < 8; ++e) {
      const int t = m0 + e + 8 * hi;
      if (kv0 + col > t)      s0[e] = -1e30f;
      if (kv0 + 16 + col > t) s1[e] = -1e30f;

      float mx = fmaxf(s0[e], s1[e]);
#pragma unroll
      for (int off = 1; off < 16; off <<= 1)
        mx = fmaxf(mx, __shfl_xor(mx, off, 32));
      const float mn = fmaxf(m[e], mx);
      const float sf = __expf(m[e] - mn);
      const float p0 = __expf(s0[e] - mn);
      const float p1 = __expf(s1[e] - mn);
      float sum = p0 + p1;
#pragma unroll
      for (int off = 1; off < 16; off <<= 1)
        sum += __shfl_xor(sum, off, 32);
      m[e] = mn;
      l[e] = l[e] * sf + sum;
#pragma unroll
      for (int j = 0; j < 8; ++j) o[j][e] *= sf;

      const int row = e + 8 * hi;
      Pl[wid][row * 32 + col]      = (bf16)p0;
      Pl[wid][row * 32 + 16 + col] = (bf16)p1;
    }

    // O += P * V : batch frag loads, then WMMAs
    v16bf pa = ldA16x32(&Pl[wid][0], 32);
    v16bf bv[4];
#pragma unroll
    for (int j = 0; j < 4; ++j) bv[j] = ldB32x16(&Vt[buf][(j * 16) * 32], 32);
#pragma unroll
    for (int j = 0; j < 4; ++j) o[j] = wmma_bf16(pa, bv[j], o[j]);
#pragma unroll
    for (int j = 0; j < 4; ++j)
      bv[j] = ldB32x16(&Vt[buf][((j + 4) * 16) * 32], 32);
#pragma unroll
    for (int j = 0; j < 4; ++j) o[j + 4] = wmma_bf16(pa, bv[j], o[j + 4]);
  }

  // normalize and write y[b, t, h*128 + d] as bf16
  bf16* ybase = yb + (size_t)b * T_ * (NH * HD) + h * HD;
#pragma unroll
  for (int j = 0; j < 8; ++j) {
    const int d = j * 16 + col;
#pragma unroll
    for (int e = 0; e < 8; ++e) {
      const int t = m0 + e + 8 * hi;
      ybase[(size_t)t * (NH * HD) + d] = (bf16)(o[j][e] / l[e]);
    }
  }
}

// ---- Kernel 3: output projection, fp32 result ------------------------------
__global__ void __launch_bounds__(128)
proj_kernel(const bf16* __restrict__ yb, const bf16* __restrict__ wpb,
            float* __restrict__ out) {
  __shared__ __align__(32) bf16 Bt[2][128 * 32];

  const int wid  = threadIdx.x >> 5;
  const int lane = threadIdx.x & 31;
  const int m0 = blockIdx.x * 64 + wid * 16;
  const int n0 = blockIdx.y * 128;

  const bf16* aBase = yb  + (size_t)m0 * C_;
  const bf16* bBase = wpb + (size_t)n0 * C_;

  auto stageB = [&](int k0, int buf) {
#pragma unroll
    for (int it = 0; it < 4; ++it) {
      const int chunk = threadIdx.x + 128 * it;
      const int row = chunk >> 2;
      const int off = (chunk & 3) * 8;
      async_copy16(&Bt[buf][row * 32 + off],
                   bBase + (size_t)row * C_ + k0 + off);
    }
  };

  const v8f zero = {0.f,0.f,0.f,0.f,0.f,0.f,0.f,0.f};
  v8f acc[8];
#pragma unroll
  for (int j = 0; j < 8; ++j) acc[j] = zero;

  stageB(0, 0);
  for (int k0 = 0; k0 < C_; k0 += 32) {
    const int buf = (k0 >> 5) & 1;
    wait_async0();
    __syncthreads();
    if (k0 + 32 < C_) stageB(k0 + 32, buf ^ 1);
    if (k0 + 32 < C_) __builtin_prefetch(aBase + k0 + 32, 0, 0);
    v16bf a = ldA16x32(aBase + k0, C_);
    v16bf bfr[8];
#pragma unroll
    for (int j = 0; j < 8; ++j) bfr[j] = ldB32x16(&Bt[buf][(j * 16) * 32], 32);
#pragma unroll
    for (int j = 0; j < 8; ++j) acc[j] = wmma_bf16(a, bfr[j], acc[j]);
  }

  const int hi = lane >> 4, col = lane & 15;
#pragma unroll
  for (int j = 0; j < 8; ++j)
#pragma unroll
    for (int e = 0; e < 8; ++e)
      out[(size_t)(m0 + e + 8 * hi) * C_ + n0 + j * 16 + col] = acc[j][e];
}

// ---------------------------------------------------------------------------
extern "C" void kernel_launch(void* const* d_in, const int* in_sizes, int n_in,
                              void* d_out, int out_size, void* d_ws,
                              size_t ws_size, hipStream_t stream) {
  (void)in_sizes; (void)n_in; (void)out_size; (void)ws_size;
  const float* x      = (const float*)d_in[0];
  const float* w_attn = (const float*)d_in[1];
  const float* w_proj = (const float*)d_in[2];
  const float* cosb   = (const float*)d_in[3];
  const float* sinb   = (const float*)d_in[4];
  float* out = (float*)d_out;

  char* ws = (char*)d_ws;
  size_t off = 0;
  bf16* xb  = (bf16*)(ws + off); off += (size_t)B_ * T_ * C_ * 2;       // 16 MB
  bf16* wab = (bf16*)(ws + off); off += (size_t)F_ * C_ * 2;            // 12 MB
  bf16* wpb = (bf16*)(ws + off); off += (size_t)C_ * (NH * HD) * 2;     //  8 MB
  bf16* qb  = (bf16*)(ws + off); off += (size_t)B_ * NH * T_ * HD * 2;  // 16 MB
  bf16* kb  = (bf16*)(ws + off); off += (size_t)B_ * NG * T_ * HD * 2;  //  4 MB
  bf16* vtb = (bf16*)(ws + off); off += (size_t)B_ * NG * HD * T_ * 2;  //  4 MB
  bf16* yb  = (bf16*)(ws + off); off += (size_t)B_ * T_ * NH * HD * 2;  // 16 MB

  int n;
  n = B_ * T_ * C_;
  cvt_f32_bf16<<<n / 256, 256, 0, stream>>>(x, xb, n);
  n = F_ * C_;
  cvt_f32_bf16<<<n / 256, 256, 0, stream>>>(w_attn, wab, n);
  n = C_ * NH * HD;
  cvt_f32_bf16<<<n / 256, 256, 0, stream>>>(w_proj, wpb, n);

  qkv_rope_kernel<<<dim3((B_ * T_) / 64, NSLOT), 128, 0, stream>>>(
      xb, wab, cosb, sinb, qb, kb, vtb);

  attn_kernel<<<dim3(T_ / 64, NH, B_), 128, 0, stream>>>(qb, kb, vtb, yb);

  proj_kernel<<<dim3((B_ * T_) / 64, (NH * HD) / 128), 128, 0, stream>>>(
      yb, wpb, out);
}
